// MultiQueryAttention_31817117728997
// MI455X (gfx1250) — matrix operand order repaired and checked by reading, not compiled, not run on hardware
//
#include <hip/hip_runtime.h>
#include <hip/hip_bf16.h>
#include <math.h>

// ---------------------------------------------------------------------------
// CDNA5 / gfx1250 MQA. bf16 WMMA everywhere (v_wmma_f32_16x16x32_bf16),
// single up-front fp32->bf16 conversion, contiguous b128 fragment loads,
// async global->LDS staging (ASYNCcnt) for shared GEMM operands.
// ---------------------------------------------------------------------------

typedef __attribute__((ext_vector_type(16))) __bf16 v16bf;
typedef __attribute__((ext_vector_type(8)))  float  v8f;

union Frag16 {            // two contiguous 16B runs -> one 16x32 bf16 fragment
    uint4 u[2];
    v16bf f;
};

__device__ __forceinline__ v8f wmma_bf16(v16bf a, v16bf b, v8f c) {
    return __builtin_amdgcn_wmma_f32_16x16x32_bf16(
        false, a, false, b, (short)0, c, false, false);
}

// Async global -> LDS copy of 16 bytes per lane (GLOBAL_LOAD_ASYNC_TO_LDS_B128,
// tracked by ASYNCcnt). lds_off is the byte offset in LDS (hardware uses
// addr[31:0] of the shared aperture, so truncating the flat address is exact).
__device__ __forceinline__ void async_b128_to_lds(unsigned lds_off, const void* g) {
    asm volatile("global_load_async_to_lds_b128 %0, %1, off"
                 :: "v"(lds_off), "v"(g) : "memory");
}
__device__ __forceinline__ void wait_async0() {
    asm volatile("s_wait_asynccnt 0" ::: "memory");
}

// ---------------------------------------------------------------------------
// Pre-pass: fp32 -> bf16 elementwise (8 elems/thread)
// ---------------------------------------------------------------------------
__global__ void __launch_bounds__(256)
cvt_bf16_kernel(const float* __restrict__ X, __bf16* __restrict__ Y) {
    size_t i = ((size_t)blockIdx.x * 256 + threadIdx.x) * 8;
    float4 a = *(const float4*)(X + i);
    float4 b = *(const float4*)(X + i + 4);
    union { __bf16 h[8]; uint4 u; } pk;
    pk.h[0] = (__bf16)a.x; pk.h[1] = (__bf16)a.y;
    pk.h[2] = (__bf16)a.z; pk.h[3] = (__bf16)a.w;
    pk.h[4] = (__bf16)b.x; pk.h[5] = (__bf16)b.y;
    pk.h[6] = (__bf16)b.z; pk.h[7] = (__bf16)b.w;
    *(uint4*)(Y + i) = pk.u;
}

// Pre-pass: W[K][N] fp32 -> WT[N][K] bf16
__global__ void __launch_bounds__(256)
transpose_cvt_kernel(const float* __restrict__ W, __bf16* __restrict__ WT,
                     int K, int N) {
    size_t idx = (size_t)blockIdx.x * 256 + threadIdx.x;   // over N*K
    int n = (int)(idx / K);
    int k = (int)(idx % K);
    WT[idx] = (__bf16)W[(size_t)k * N + n];
}

// ---------------------------------------------------------------------------
// GEMM: Y[M,N] = A[M,K](bf16) @ WT[N,K](bf16)^T + bias.
// 4 waves/block; block tile 128x64; wave tile 32x64 (8 WMMAs / 32-K step).
// WT tile staged to LDS with async b128 copies.
// ---------------------------------------------------------------------------
template <bool BF16_OUT, bool TRANS_OUT>
__global__ void __launch_bounds__(128)
gemm_abT_wmma(const __bf16* __restrict__ A, const __bf16* __restrict__ WT,
              const float* __restrict__ bias, void* __restrict__ Yv,
              int M, int N, int K) {
    __shared__ __bf16 ldsB[64 * 32];           // 64 N-rows x 32 K, 4 KB

    const int nBlocksN = N >> 6;
    const int gm = blockIdx.x / nBlocksN;
    const int gn = blockIdx.x % nBlocksN;
    const int t    = threadIdx.x;
    const int wave = t >> 5;
    const int lane = t & 31;
    const int half = lane >> 4;
    const int mn   = lane & 15;

    const int rbase = gm * 128 + wave * 32;

    // staging assignment: 2 threads per WT row, 16 bf16 each (2 async b128)
    const int sn  = t >> 1;                    // 0..63
    const int seg = (t & 1) * 16;              // 0 or 16 elements

    v8f acc[2][4];
#pragma unroll
    for (int mt = 0; mt < 2; ++mt)
#pragma unroll
        for (int nt = 0; nt < 4; ++nt) acc[mt][nt] = v8f{};

    for (int k0 = 0; k0 < K; k0 += 32) {
        // ---- stage WT tile (shared by all 4 waves) into LDS, async ----
        const __bf16* gsrc = WT + (size_t)(gn * 64 + sn) * K + k0 + seg;
        unsigned loff = (unsigned)(size_t)&ldsB[sn * 32 + seg];
        async_b128_to_lds(loff, gsrc);
        async_b128_to_lds(loff + 16, gsrc + 8);
        wait_async0();
        __syncthreads();

        // ---- A fragments: contiguous b128 pairs straight from global ----
        Frag16 fa[2];
#pragma unroll
        for (int mt = 0; mt < 2; ++mt) {
            const __bf16* ap = A + (size_t)(rbase + mt * 16 + mn) * K + k0;
            fa[mt].u[0] = *(const uint4*)(ap + 16 * half);
            fa[mt].u[1] = *(const uint4*)(ap + 16 * half + 8);
        }
        // ---- B fragments from LDS + 8 WMMAs ----
#pragma unroll
        for (int nt = 0; nt < 4; ++nt) {
            Frag16 fb;
            const __bf16* bp = &ldsB[(nt * 16 + mn) * 32 + 16 * half];
            fb.u[0] = *(const uint4*)(bp);
            fb.u[1] = *(const uint4*)(bp + 8);
#pragma unroll
            for (int mt = 0; mt < 2; ++mt)
                acc[mt][nt] = wmma_bf16(fa[mt].f, fb.f, acc[mt][nt]);
        }
        __syncthreads();
    }

    // ---- epilogue: bias + store ----
    const int ncol = gn * 64;
#pragma unroll
    for (int nt = 0; nt < 4; ++nt) {
        const int n = ncol + nt * 16 + mn;
        const float bn = bias[n];
#pragma unroll
        for (int mt = 0; mt < 2; ++mt) {
#pragma unroll
            for (int v = 0; v < 8; ++v) {
                const int m = rbase + mt * 16 + v + 8 * half;
                const float val = acc[mt][nt][v] + bn;
                if (TRANS_OUT) {
                    ((__bf16*)Yv)[(size_t)n * M + m] = (__bf16)val;
                } else if (BF16_OUT) {
                    ((__bf16*)Yv)[(size_t)m * N + n] = (__bf16)val;
                } else {
                    ((float*)Yv)[(size_t)m * N + n] = val;
                }
            }
        }
    }
}

// ---------------------------------------------------------------------------
// MQA flash attention, bf16 operands. One wave per (b, h, 16-row Q tile).
// Qb: [B*S][E] bf16 (head h at cols h*64..); Kb: [B*S][64] bf16;
// Vt: [64][B*S] bf16 (transposed); Ob: [B*S][E] bf16.
// ---------------------------------------------------------------------------
__global__ void __launch_bounds__(32)
mqa_attention_wmma(const __bf16* __restrict__ Qb, const __bf16* __restrict__ Kb,
                   const __bf16* __restrict__ Vt, __bf16* __restrict__ Ob,
                   int Bn, int S, int H, int E) {
    const int qtiles = S >> 4;
    int idx = blockIdx.x;
    const int qt = idx % qtiles; idx /= qtiles;
    const int h  = idx % H;
    const int b  = idx / H;
    const int M  = Bn * S;

    const int lane = threadIdx.x;
    const int half = lane >> 4;
    const int mn   = lane & 15;

    __shared__ __bf16 pLDS[16 * 32];     // P tile, row-major 16x32 bf16

    const __bf16* Qbase = Qb + ((size_t)b * S) * E + h * 64;
    const __bf16* Kbase = Kb + ((size_t)b * S) * 64;
    const __bf16* Vbase = Vt + (size_t)b * S;          // row stride M

    // Q tile 16x64 -> two A fragments, pure b128 loads
    Frag16 aq[2];
#pragma unroll
    for (int ks = 0; ks < 2; ++ks) {
        const __bf16* qp = Qbase + (size_t)(qt * 16 + mn) * E + ks * 32;
        aq[ks].u[0] = *(const uint4*)(qp + 16 * half);
        aq[ks].u[1] = *(const uint4*)(qp + 16 * half + 8);
    }

    float mrow[8], lrow[8];
    v8f acc[4] = {v8f{}, v8f{}, v8f{}, v8f{}};
#pragma unroll
    for (int v = 0; v < 8; ++v) { mrow[v] = -__builtin_inff(); lrow[v] = 0.0f; }

    const float scale = 0.125f;          // 1/sqrt(64)

    for (int j0 = 0; j0 < S; j0 += 32) {
        // ---- scores: S = Q @ K^T, two 16x16 tiles ----
        v8f sc[2] = {v8f{}, v8f{}};
#pragma unroll
        for (int nt = 0; nt < 2; ++nt) {
            const __bf16* kp = Kbase + (size_t)(j0 + nt * 16 + mn) * 64;
#pragma unroll
            for (int ks = 0; ks < 2; ++ks) {
                Frag16 bk;
                bk.u[0] = *(const uint4*)(kp + ks * 32 + 16 * half);
                bk.u[1] = *(const uint4*)(kp + ks * 32 + 16 * half + 8);
                sc[nt] = wmma_bf16(aq[ks].f, bk.f, sc[nt]);
            }
        }

        // ---- online softmax; row m = v + 8*half, 16-lane reductions ----
        float p0[8], p1[8], corr[8];
#pragma unroll
        for (int v = 0; v < 8; ++v) {
            float s0 = sc[0][v] * scale;
            float s1 = sc[1][v] * scale;
            float rmax = fmaxf(s0, s1);
#pragma unroll
            for (int off = 8; off >= 1; off >>= 1)
                rmax = fmaxf(rmax, __shfl_xor(rmax, off));
            float newm = fmaxf(mrow[v], rmax);
            corr[v] = __expf(mrow[v] - newm);
            mrow[v] = newm;
            p0[v] = __expf(s0 - newm);
            p1[v] = __expf(s1 - newm);
            float rsum = p0[v] + p1[v];
#pragma unroll
            for (int off = 8; off >= 1; off >>= 1)
                rsum += __shfl_xor(rsum, off);
            lrow[v] = lrow[v] * corr[v] + rsum;
        }
#pragma unroll
        for (int t = 0; t < 4; ++t)
#pragma unroll
            for (int v = 0; v < 8; ++v)
                acc[t][v] *= corr[v];

        // ---- P: C-layout -> A-layout via LDS (bf16) ----
        __syncthreads();
#pragma unroll
        for (int v = 0; v < 8; ++v) {
            int m = v + 8 * half;
            pLDS[m * 32 + mn]      = (__bf16)p0[v];
            pLDS[m * 32 + 16 + mn] = (__bf16)p1[v];
        }
        __syncthreads();

        Frag16 pa;
        pa.u[0] = *(const uint4*)&pLDS[mn * 32 + 16 * half];
        pa.u[1] = *(const uint4*)&pLDS[mn * 32 + 16 * half + 8];
        __syncthreads();

        // ---- acc += P(16x32) @ V(32x64); V^T rows are contiguous keys ----
#pragma unroll
        for (int nt = 0; nt < 4; ++nt) {
            Frag16 bv;
            const __bf16* vp = Vbase + (size_t)(nt * 16 + mn) * M + j0 + 16 * half;
            bv.u[0] = *(const uint4*)(vp);
            bv.u[1] = *(const uint4*)(vp + 8);
            acc[nt] = wmma_bf16(pa.f, bv.f, acc[nt]);
        }
    }

    // ---- normalize, store bf16 O ----
#pragma unroll
    for (int nt = 0; nt < 4; ++nt) {
#pragma unroll
        for (int v = 0; v < 8; ++v) {
            int row = qt * 16 + v + 8 * half;
            int d   = nt * 16 + mn;
            Ob[((size_t)b * S + row) * E + h * 64 + d] =
                (__bf16)(acc[nt][v] / lrow[v]);
        }
    }
}

// ---------------------------------------------------------------------------
extern "C" void kernel_launch(void* const* d_in, const int* in_sizes, int n_in,
                              void* d_out, int out_size, void* d_ws, size_t ws_size,
                              hipStream_t stream) {
    (void)in_sizes; (void)n_in; (void)out_size; (void)ws_size;

    const float* x  = (const float*)d_in[0];
    const float* wq = (const float*)d_in[1];
    const float* bq = (const float*)d_in[2];
    const float* wk = (const float*)d_in[3];
    const float* bk = (const float*)d_in[4];
    const float* wv = (const float*)d_in[5];
    const float* bv = (const float*)d_in[6];
    const float* wo = (const float*)d_in[7];
    const float* bo = (const float*)d_in[8];
    float* out = (float*)d_out;

    const int Bn = 2, S = 2048, E = 1024, H = 16, D = 64;
    const int M = Bn * S;                       // 4096

    // Workspace layout (bf16): ~31 MB
    __bf16* xb  = (__bf16*)d_ws;                // M*E
    __bf16* wqT = xb  + (size_t)M * E;          // E*E
    __bf16* wkT = wqT + (size_t)E * E;          // D*E
    __bf16* wvT = wkT + (size_t)D * E;          // D*E
    __bf16* woT = wvT + (size_t)D * E;          // E*E
    __bf16* Qb  = woT + (size_t)E * E;          // M*E
    __bf16* Kb  = Qb  + (size_t)M * E;          // M*D
    __bf16* Vt  = Kb  + (size_t)M * D;          // D*M (transposed)
    __bf16* Ob  = Vt  + (size_t)D * M;          // M*E

    // --- conversions / transposes (once) ---
    cvt_bf16_kernel<<<dim3((unsigned)((size_t)M * E / 8 / 256)), dim3(256), 0, stream>>>(x, xb);
    transpose_cvt_kernel<<<dim3((unsigned)((size_t)E * E / 256)), dim3(256), 0, stream>>>(wq, wqT, E, E);
    transpose_cvt_kernel<<<dim3((unsigned)((size_t)D * E / 256)), dim3(256), 0, stream>>>(wk, wkT, E, D);
    transpose_cvt_kernel<<<dim3((unsigned)((size_t)D * E / 256)), dim3(256), 0, stream>>>(wv, wvT, E, D);
    transpose_cvt_kernel<<<dim3((unsigned)((size_t)E * E / 256)), dim3(256), 0, stream>>>(wo, woT, E, E);

    dim3 blk(128);
    // Projections (bf16 out; V stored transposed [D][M])
    gemm_abT_wmma<true, false><<<dim3((M / 128) * (E / 64)), blk, 0, stream>>>(xb, wqT, bq, Qb, M, E, E);
    gemm_abT_wmma<true, false><<<dim3((M / 128) * (D / 64)), blk, 0, stream>>>(xb, wkT, bk, Kb, M, D, E);
    gemm_abT_wmma<true, true ><<<dim3((M / 128) * (D / 64)), blk, 0, stream>>>(xb, wvT, bv, Vt, M, D, E);

    // Attention
    mqa_attention_wmma<<<dim3(Bn * H * (S / 16)), dim3(32), 0, stream>>>(Qb, Kb, Vt, Ob, Bn, S, H, E);

    // Output projection (fp32 out + bias)
    gemm_abT_wmma<false, false><<<dim3((M / 128) * (E / 64)), blk, 0, stream>>>(Ob, woT, bo, out, M, E, E);
}
